// S6_15607911154022
// MI455X (gfx1250) — matrix operand order repaired and checked by reading, not compiled, not run on hardware
//
#include <hip/hip_runtime.h>

typedef __attribute__((ext_vector_type(16))) __bf16 v16bf;
typedef __attribute__((ext_vector_type(8)))  __bf16 v8bf;
typedef __attribute__((ext_vector_type(4)))  __bf16 v4bf;
typedef __attribute__((ext_vector_type(8)))  float  v8f;

#define DDIM 1024
#define BM 128
#define BN 128
#define BK 32
#define LDT 40   // padded LDS row stride in u16 units (80 B, 16B aligned, bank-conflict friendly)

static __device__ __forceinline__ float softplus_f(float z) {
    return fmaxf(z, 0.0f) + log1pf(expf(-fabsf(z)));
}
static __device__ __forceinline__ v16bf cat8(v8bf lo, v8bf hi) {
    return __builtin_shufflevector(lo, hi, 0,1,2,3,4,5,6,7,8,9,10,11,12,13,14,15);
}

// CDNA5 async memory->LDS copy (ASYNCcnt-tracked), inline asm for toolchain portability.
static __device__ __forceinline__ void async_copy_b128(unsigned lds_addr, const void* gaddr) {
    asm volatile("global_load_async_to_lds_b128 %0, %1, off"
                 :: "v"(lds_addr), "v"(gaddr) : "memory");
}
static __device__ __forceinline__ void wait_async0() {
    asm volatile("s_wait_asynccnt 0x0" ::: "memory");
}

// ---------------- fp32 -> bf16 convert of x ----------------
__global__ void __launch_bounds__(256) k_cvt_x(const float* __restrict__ in,
                                               __bf16* __restrict__ out, int n4) {
    int i = blockIdx.x * blockDim.x + threadIdx.x;
    if (i >= n4) return;
    float4 v = reinterpret_cast<const float4*>(in)[i];
    v4bf o = { (__bf16)v.x, (__bf16)v.y, (__bf16)v.z, (__bf16)v.w };
    reinterpret_cast<v4bf*>(out)[i] = o;
}

// ------------- W1 [k][n] fp32 -> w1t [n][k] bf16 (LDS tile transpose) -------------
__global__ void __launch_bounds__(256) k_cvt_w1t(const float* __restrict__ W1,
                                                 __bf16* __restrict__ w1t) {
    __shared__ float tile[32][33];
    const int kt = blockIdx.y * 32, nt = blockIdx.x * 32;
    const int tx = threadIdx.x, ty = threadIdx.y; // 32 x 8
    #pragma unroll
    for (int i = 0; i < 32; i += 8)
        tile[ty + i][tx] = W1[(size_t)(kt + ty + i) * DDIM + nt + tx];
    __syncthreads();
    #pragma unroll
    for (int i = 0; i < 32; i += 8)
        w1t[(size_t)(nt + ty + i) * DDIM + kt + tx] = (__bf16)tile[tx][ty + i];
}

// ------------- [W2|W3] fp32 [k][16] -> w23t [32][k] bf16 (transposed, packed) -------------
__global__ void __launch_bounds__(256) k_cvt_w23t(const float* __restrict__ W2,
                                                  const float* __restrict__ W3,
                                                  __bf16* __restrict__ w23t) {
    int id = blockIdx.x * blockDim.x + threadIdx.x; // 32 * 1024
    int n = id >> 10, k = id & 1023;
    float v = (n < 16) ? W2[k * 16 + n] : W3[k * 16 + (n - 16)];
    w23t[id] = (__bf16)v;
}

// ------------- s[row] = sum_n (x@W2+b2)*(x@W3+b3) via two WMMA chains -------------
// block = 128 threads (4 waves), each wave owns 16 rows.
__global__ void __launch_bounds__(128) k_bc_s(const __bf16* __restrict__ xh,
                                              const __bf16* __restrict__ w23t,
                                              const float* __restrict__ b2,
                                              const float* __restrict__ b3,
                                              float* __restrict__ s) {
    const int wave = threadIdx.x >> 5;
    const int lane = threadIdx.x & 31;
    const int hf = lane >> 4, r = lane & 15;
    const int m0 = (blockIdx.x * 4 + wave) * 16;

    v8f acc0 = {}, acc1 = {};
    // A fragment: interleaved layout -> 16B loads at +16*hf and +32 more
    const char* aBase = (const char*)xh + (size_t)(m0 + r) * (DDIM * 2) + hf * 16;
    // B fragments: interleaved layout -> 16B loads at +16*hf and +32 more
    const char* b0Base = (const char*)w23t + (size_t)r * (DDIM * 2) + hf * 16;
    const char* b1Base = (const char*)w23t + (size_t)(16 + r) * (DDIM * 2) + hf * 16;

    for (int kk = 0; kk < DDIM; kk += BK) {
        v16bf a  = cat8(*(const v8bf*)(aBase + kk * 2),
                        *(const v8bf*)(aBase + kk * 2 + 32));
        v16bf bA = cat8(*(const v8bf*)(b0Base + kk * 2),
                        *(const v8bf*)(b0Base + kk * 2 + 32));
        v16bf bB = cat8(*(const v8bf*)(b1Base + kk * 2),
                        *(const v8bf*)(b1Base + kk * 2 + 32));
        acc0 = __builtin_amdgcn_wmma_f32_16x16x32_bf16(false, a, false, bA,
                                                       (short)0, acc0, false, false);
        acc1 = __builtin_amdgcn_wmma_f32_16x16x32_bf16(false, a, false, bB,
                                                       (short)0, acc1, false, false);
    }
    const float bb2 = b2[r], bb3 = b3[r];
    #pragma unroll
    for (int v = 0; v < 8; ++v) {
        float p = (acc0[v] + bb2) * (acc1[v] + bb3);
        p += __shfl_xor(p, 1);
        p += __shfl_xor(p, 2);
        p += __shfl_xor(p, 4);
        p += __shfl_xor(p, 8);
        if (r == 0) s[m0 + v + 8 * hf] = p;   // C-layout: row = v + 8*half
    }
}

// ------------- main fused GEMM: y = x * softplus(X@W1 + b1) * s[row] -------------
// 256 threads (8 waves); 128x128 tile, BK=32; double-buffered LDS fed by
// global_load_async_to_lds_b128; one barrier per K-step; wave -> 32x64 sub-tile.
__global__ void __launch_bounds__(256) k_main(const __bf16* __restrict__ xh,
                                              const __bf16* __restrict__ w1t,
                                              const float* __restrict__ x,
                                              const float* __restrict__ bias1,
                                              const float* __restrict__ s,
                                              float* __restrict__ y) {
    __shared__ __align__(16) __bf16 Asm[2][BM * LDT];
    __shared__ __align__(16) __bf16 Bsm[2][BN * LDT];

    const int tid = threadIdx.x;
    const int wave = tid >> 5, lane = tid & 31;
    const int hf = lane >> 4, r = lane & 15;
    const int moff = (wave >> 1) * 32;   // 4 waves along M
    const int noff = (wave & 1) * 64;    // 2 waves along N
    const size_t rowM = (size_t)blockIdx.y * BM;
    const int colN = blockIdx.x * BN;

    v8f acc[2][4] = {};

    // cooperative tile fetch: 512 16B chunks per matrix, 2 per thread (rows rrow, rrow+64)
    const int rrow = tid >> 2, seg = tid & 3;
    const char* gA = (const char*)xh  + (rowM + rrow) * (size_t)(DDIM * 2) + seg * 16;
    const char* gB = (const char*)w1t + (size_t)(colN + rrow) * (DDIM * 2) + seg * 16;
    const size_t gSkip = (size_t)64 * (DDIM * 2);
    const int lSkip = 64 * (LDT * 2);

    unsigned lA0[2], lB0[2];
    const char* pA0[2];
    const char* pB0[2];
    #pragma unroll
    for (int b = 0; b < 2; ++b) {
        lA0[b] = (unsigned)(uintptr_t)((const char*)&Asm[b][0] + rrow * (LDT * 2) + seg * 16);
        lB0[b] = (unsigned)(uintptr_t)((const char*)&Bsm[b][0] + rrow * (LDT * 2) + seg * 16);
        pA0[b] = (const char*)&Asm[b][0] + (moff + r) * (LDT * 2) + hf * 16;
        pB0[b] = (const char*)&Bsm[b][0] + (noff + r) * (LDT * 2) + hf * 16;
    }

    // prime buffer 0
    async_copy_b128(lA0[0],         gA);
    async_copy_b128(lA0[0] + lSkip, gA + gSkip);
    async_copy_b128(lB0[0],         gB);
    async_copy_b128(lB0[0] + lSkip, gB + gSkip);

    int buf = 0;
    for (int kk = 0; kk < DDIM; kk += BK, buf ^= 1) {
        wait_async0();       // our async writes into buf are done
        __syncthreads();     // everyone's writes visible; prev buf fully consumed

        if (kk + BK < DDIM) {            // overlap next tile fetch with WMMA
            const char* pa = gA + (size_t)(kk + BK) * 2;
            const char* pb = gB + (size_t)(kk + BK) * 2;
            async_copy_b128(lA0[buf ^ 1],         pa);
            async_copy_b128(lA0[buf ^ 1] + lSkip, pa + gSkip);
            async_copy_b128(lB0[buf ^ 1],         pb);
            async_copy_b128(lB0[buf ^ 1] + lSkip, pb + gSkip);
            if (kk + 2 * BK < DDIM) {    // L2 warm-up one more tile ahead
                __builtin_prefetch(gA + (size_t)(kk + 2 * BK) * 2, 0, 0);
                __builtin_prefetch(gB + (size_t)(kk + 2 * BK) * 2, 0, 0);
            }
        }

        v16bf af[2], bfr[4];
        #pragma unroll
        for (int mi = 0; mi < 2; ++mi) {
            const char* p = pA0[buf] + mi * 16 * (LDT * 2);
            af[mi] = cat8(*(const v8bf*)p, *(const v8bf*)(p + 32));
        }
        #pragma unroll
        for (int ni = 0; ni < 4; ++ni) {
            const char* p = pB0[buf] + ni * 16 * (LDT * 2);
            bfr[ni] = cat8(*(const v8bf*)p, *(const v8bf*)(p + 32));
        }
        #pragma unroll
        for (int ni = 0; ni < 4; ++ni) {
            acc[0][ni] = __builtin_amdgcn_wmma_f32_16x16x32_bf16(
                false, af[0], false, bfr[ni], (short)0, acc[0][ni], false, false);
            acc[1][ni] = __builtin_amdgcn_wmma_f32_16x16x32_bf16(
                false, af[1], false, bfr[ni], (short)0, acc[1][ni], false, false);
        }
    }

    // fused epilogue: y = x * softplus(acc + b1) * s[row]
    #pragma unroll
    for (int mi = 0; mi < 2; ++mi) {
        const size_t growBase = rowM + moff + mi * 16 + 8 * hf;
        #pragma unroll
        for (int v = 0; v < 8; ++v) {
            const size_t grow = growBase + v;
            const float sv = s[grow];
            #pragma unroll
            for (int ni = 0; ni < 4; ++ni) {
                const int gcol = colN + noff + ni * 16 + r;
                const float z = acc[mi][ni][v] + bias1[gcol];
                y[grow * DDIM + gcol] = x[grow * DDIM + gcol] * softplus_f(z) * sv;
            }
        }
    }
}

extern "C" void kernel_launch(void* const* d_in, const int* in_sizes, int n_in,
                              void* d_out, int out_size, void* d_ws, size_t ws_size,
                              hipStream_t stream) {
    (void)n_in; (void)out_size; (void)ws_size;
    const float* x  = (const float*)d_in[0];
    const float* W1 = (const float*)d_in[1];
    const float* b1 = (const float*)d_in[2];
    const float* W2 = (const float*)d_in[3];
    const float* b2 = (const float*)d_in[4];
    const float* W3 = (const float*)d_in[5];
    const float* b3 = (const float*)d_in[6];
    // d_in[7] = A is unused: it multiplies h == 0 in the reference.
    float* y = (float*)d_out;

    const int rows = in_sizes[0] / DDIM;   // b*l = 4096

    char* ws = (char*)d_ws;
    size_t off = 0;
    __bf16* xh = (__bf16*)(ws + off);
    off += (size_t)rows * DDIM * 2;  off = (off + 255) & ~(size_t)255;
    __bf16* w1t = (__bf16*)(ws + off);
    off += (size_t)DDIM * DDIM * 2;  off = (off + 255) & ~(size_t)255;
    __bf16* w23t = (__bf16*)(ws + off);
    off += (size_t)32 * DDIM * 2;    off = (off + 255) & ~(size_t)255;
    float* s = (float*)(ws + off);

    k_cvt_x<<<(rows * DDIM / 4 + 255) / 256, 256, 0, stream>>>(x, xh, rows * DDIM / 4);
    k_cvt_w1t<<<dim3(DDIM / 32, DDIM / 32), dim3(32, 8), 0, stream>>>(W1, w1t);
    k_cvt_w23t<<<(32 * DDIM) / 256, 256, 0, stream>>>(W2, W3, w23t);
    k_bc_s<<<rows / 64, 128, 0, stream>>>(xh, w23t, b2, b3, s);
    k_main<<<dim3(DDIM / BN, rows / BM), 256, 0, stream>>>(xh, w1t, x, b1, s, y);
}
